// EBMColorsGraph_46196668236125
// MI455X (gfx1250) — compile-verified, matrix operand order checked by reading, not run on hardware
//
#include <hip/hip_runtime.h>
#include <hip/hip_bf16.h>
#include <stdint.h>

#define BB 32
#define NN 4096
#define CC 8
#define EE 16384
#define HH 128
#define TILE_M 128

typedef __attribute__((ext_vector_type(16))) __bf16 v16bf;
typedef __attribute__((ext_vector_type(8)))  float  v8f;
typedef __attribute__((ext_vector_type(4)))  unsigned int v4u;

union FragU {
    v16bf v;
    unsigned int w[8];
    v4u q[2];
};

// Native f32 -> bf16 conversion (hardware cvt, RNE).
__device__ __forceinline__ unsigned short f2bf(float f) {
    __bf16 h = (__bf16)f;
    return __builtin_bit_cast(unsigned short, h);
}
__device__ __forceinline__ unsigned int pack2bf(float lo, float hi) {
    return (unsigned int)f2bf(lo) | ((unsigned int)f2bf(hi) << 16);
}
// Fast sigmoid: v_exp_f32 + v_rcp_f32 (avoid IEEE div_scale/div_fmas sequence).
__device__ __forceinline__ float sigmoidf_(float z) {
    return __builtin_amdgcn_rcpf(1.0f + __expf(-z));
}

// K offset inside a 16-bit A/B fragment for VGPR v (0..7), lane-half h (0/1).
// CDNA5 ISA: VGPR0..3 hold K = 8h + {0,1},{2,3},{4,5},{6,7};
//                 VGPR4..7 hold K = 16 + 8h + {0,1},...   (always even base)
__device__ __forceinline__ int frag_k(int v, int h) {
    return (v < 4) ? (8 * h + 2 * v) : (16 + 8 * h + 2 * (v - 4));
}

// A fragment from row-major [m][k] bf16 LDS; rs multiple of 8 -> two b128 loads.
__device__ __forceinline__ v16bf load_frag_A(const unsigned short* base, int rs,
                                             int mBase, int kBase, int lane) {
    const int h = lane >> 4;
    const int m = mBase + (lane & 15);
    const unsigned short* p = base + m * rs + kBase + 8 * h;
    FragU f;
    f.q[0] = *(const v4u*)(p);        // k = {0..7} within half
    f.q[1] = *(const v4u*)(p + 16);   // k = {16..23} within half
    return f.v;
}

// B fragment from pair-packed [k/2][n] uint LDS (pairs along k): one b32 per VGPR.
__device__ __forceinline__ v16bf load_frag_Bp(const unsigned int* base, int rs,
                                              int kBase, int nBase, int lane) {
    const int h = lane >> 4;
    const int n = nBase + (lane & 15);
    FragU f;
#pragma unroll
    for (int v = 0; v < 8; ++v) {
        int kp = (kBase + frag_k(v, h)) >> 1;
        f.w[v] = base[kp * rs + n];
    }
    return f.v;
}

// B^T fragment from pair-packed [n][k/2] uint LDS: one b32 per VGPR.
__device__ __forceinline__ v16bf load_frag_Bq(const unsigned int* base, int rs,
                                              int kBase, int nBase, int lane) {
    const int h = lane >> 4;
    const int n = nBase + (lane & 15);
    FragU f;
#pragma unroll
    for (int v = 0; v < 8; ++v) {
        int kp = (kBase + frag_k(v, h)) >> 1;
        f.w[v] = base[n * rs + kp];
    }
    return f.v;
}

__global__ void zero_kernel(float* __restrict__ p, int n) {
    int i = blockIdx.x * blockDim.x + threadIdx.x;
    if (i < n) p[i] = 0.0f;
}

__global__ __launch_bounds__(256) void ebm_grad_kernel(
    const float* __restrict__ x,   const int* __restrict__ t,
    const int* __restrict__ xi,    const float* __restrict__ ac,
    const float* __restrict__ W1,  const float* __restrict__ b1,
    const float* __restrict__ W2,  const float* __restrict__ b2,
    const float* __restrict__ W3,  float* __restrict__ out)
{
    // Pair-packed weights (uint = {bf16 k, bf16 k+1}), both orientations.
    __shared__ __align__(16) unsigned int   sW1p[16][130];  // W1 pairs along din-k (padded to 32)
    __shared__ __align__(16) unsigned int   sW2p[64][130];  // W2 pairs along k   (forward B)
    __shared__ __align__(16) unsigned int   sW2q[128][65];  // W2 pairs along n   (W2^T B)
    __shared__ __align__(16) unsigned int   sW1q[32][65];   // W1 pairs along H   (W1^T B)
    // Activations: rs = 136 (multiple of 8 -> b128-aligned A fragments).
    __shared__ __align__(16) unsigned short sH[128 * 136];     // h1, later dz1
    __shared__ __align__(16) unsigned short sPool[128 * 136];  // input tile A, later dz2
    __shared__ float sB1[128], sB2[128], sW3v[128];            // sW3v pre-scaled by 2
    __shared__ int   sSrc[128], sTgt[128];

    const int tid  = threadIdx.x;
    const int lane = tid & 31;
    const int wave = tid >> 5;
    const int b        = blockIdx.y;
    const int tileBase = blockIdx.x * TILE_M;

    const int   tb    = t[b];
    const float scale = 1.0f - ac[tb];
    const float tfeat = (float)tb * 0.001f;

    // ---- stage weights (packed pairs) ----
    for (int i = tid; i < 16 * 128; i += 256) {          // sW1p
        int kp = i >> 7, n = i & 127;
        int k0 = 2 * kp, k1 = 2 * kp + 1;
        float lo = (k0 < 17) ? W1[k0 * HH + n] : 0.0f;
        float hi = (k1 < 17) ? W1[k1 * HH + n] : 0.0f;
        sW1p[kp][n] = pack2bf(lo, hi);
    }
    for (int i = tid; i < 64 * 128; i += 256) {          // sW2p
        int kp = i >> 7, n = i & 127;
        sW2p[kp][n] = pack2bf(W2[(2 * kp) * HH + n], W2[(2 * kp + 1) * HH + n]);
    }
    for (int i = tid; i < 128 * 64; i += 256) {          // sW2q
        int a = i >> 6, bp = i & 63;
        sW2q[a][bp] = pack2bf(W2[a * HH + 2 * bp], W2[a * HH + 2 * bp + 1]);
    }
    for (int i = tid; i < 32 * 64; i += 256) {           // sW1q
        int a = i >> 6, bp = i & 63;
        sW1q[a][bp] = (a < 17) ? pack2bf(W1[a * HH + 2 * bp], W1[a * HH + 2 * bp + 1]) : 0u;
    }
    if (tid < 128) {
        sB1[tid]  = b1[tid];
        sB2[tid]  = b2[tid];
        sW3v[tid] = 2.0f * W3[tid];          // fold the d(2*sum)/de factor in here
        int e = tileBase + tid;
        sSrc[tid] = xi[(b * EE + e) * 2 + 0];
        sTgt[tid] = xi[(b * EE + e) * 2 + 1];
    }
    __syncthreads();

    // ---- gather tile: row m = [src(0..7) | tgt(8..15) | tfeat | zeros] ----
    {
        // 256 threads <-> 128 edges x 2 nodes; 8 floats per node via 2x b128.
        int e     = tid >> 1;
        int which = tid & 1;
        int node  = which ? sTgt[e] : sSrc[e];
        const float4* xv = (const float4*)(x + (size_t)(b * NN + node) * CC);
        float4 v0 = xv[0];
        float4 v1 = xv[1];
        unsigned int* dst = (unsigned int*)&sPool[e * 136 + which * 8];
        dst[0] = pack2bf(v0.x, v0.y);
        dst[1] = pack2bf(v0.z, v0.w);
        dst[2] = pack2bf(v1.x, v1.y);
        dst[3] = pack2bf(v1.z, v1.w);
    }
    for (int i = tid; i < 128 * 8; i += 256) {           // cols 16..31 (tfeat + zero pad)
        int m = i >> 3, j = i & 7;
        unsigned int vwd = (j == 0) ? pack2bf(tfeat, 0.0f) : 0u;
        *(unsigned int*)&sPool[m * 136 + 16 + 2 * j] = vwd;
    }
    __syncthreads();

    const int mBase = wave * 16;
    const int hh = lane >> 4;
    const int ln = lane & 15;

    float dsilu1[64];   // silu'(z1) kept in VGPRs; GEMM1/GEMM3 share (m,n) mapping

    // ---- GEMM1: z1 = A @ W1 + b1 ; h1 = silu(z1) ----
    {
        v16bf a0 = load_frag_A(sPool, 136, mBase, 0, lane);
#pragma unroll
        for (int nt = 0; nt < 8; ++nt) {
            v8f acc = {};
            v16bf bf = load_frag_Bp(&sW1p[0][0], 130, 0, nt * 16, lane);
            acc = __builtin_amdgcn_wmma_f32_16x16x32_bf16(false, a0, false, bf,
                                                          (short)0, acc, false, false);
#pragma unroll
            for (int r = 0; r < 8; ++r) {
                int m = mBase + r + 8 * hh;
                int n = nt * 16 + ln;
                float z  = acc[r] + sB1[n];
                float sg = sigmoidf_(z);
                float h1 = z * sg;
                sH[m * 136 + n] = f2bf(h1);
                dsilu1[nt * 8 + r] = __builtin_fmaf(h1, 1.0f - sg, sg);  // silu'(z1)
            }
        }
    }
    __syncthreads();

    // ---- GEMM2: z2 = h1 @ W2 + b2 ; dz2 = 2*W3[n]*silu'(z2) (into sPool) ----
    {
        v16bf a[4];
#pragma unroll
        for (int kk = 0; kk < 4; ++kk)
            a[kk] = load_frag_A(sH, 136, mBase, kk * 32, lane);
        for (int nt = 0; nt < 8; ++nt) {
            v8f acc = {};
#pragma unroll
            for (int kk = 0; kk < 4; ++kk) {
                v16bf bf = load_frag_Bp(&sW2p[0][0], 130, kk * 32, nt * 16, lane);
                acc = __builtin_amdgcn_wmma_f32_16x16x32_bf16(false, a[kk], false, bf,
                                                              (short)0, acc, false, false);
            }
#pragma unroll
            for (int r = 0; r < 8; ++r) {
                int m = mBase + r + 8 * hh;
                int n = nt * 16 + ln;
                float z  = acc[r] + sB2[n];
                float sg = sigmoidf_(z);
                float h2 = z * sg;
                float ds = __builtin_fmaf(h2, 1.0f - sg, sg);   // silu'(z2)
                sPool[m * 136 + n] = f2bf(sW3v[n] * ds);        // dz2 (2x folded in)
            }
        }
    }
    __syncthreads();

    // ---- GEMM3: dh1 = dz2 @ W2^T ; dz1 = dh1 * silu'(z1) (into sH) ----
    {
        v16bf a[4];
#pragma unroll
        for (int kk = 0; kk < 4; ++kk)
            a[kk] = load_frag_A(sPool, 136, mBase, kk * 32, lane);
#pragma unroll
        for (int nt = 0; nt < 8; ++nt) {
            v8f acc = {};
#pragma unroll
            for (int kk = 0; kk < 4; ++kk) {
                v16bf bf = load_frag_Bq(&sW2q[0][0], 65, kk * 32, nt * 16, lane);
                acc = __builtin_amdgcn_wmma_f32_16x16x32_bf16(false, a[kk], false, bf,
                                                              (short)0, acc, false, false);
            }
#pragma unroll
            for (int r = 0; r < 8; ++r) {
                int m = mBase + r + 8 * hh;
                int n = nt * 16 + ln;
                sH[m * 136 + n] = f2bf(acc[r] * dsilu1[nt * 8 + r]);  // dz1
            }
        }
    }
    __syncthreads();

    // ---- GEMM4: dh[:,0:16] = dz1 @ W1^T ; scatter-add scaled grads ----
    {
        v16bf a[4];
#pragma unroll
        for (int kk = 0; kk < 4; ++kk)
            a[kk] = load_frag_A(sH, 136, mBase, kk * 32, lane);
        v8f acc = {};
#pragma unroll
        for (int kk = 0; kk < 4; ++kk) {
            v16bf bf = load_frag_Bq(&sW1q[0][0], 65, kk * 32, 0, lane);
            acc = __builtin_amdgcn_wmma_f32_16x16x32_bf16(false, a[kk], false, bf,
                                                          (short)0, acc, false, false);
        }
#pragma unroll
        for (int r = 0; r < 8; ++r) {
            int m = mBase + r + 8 * hh;   // local edge index
            int f = ln;                   // 0..7 -> src grad, 8..15 -> tgt grad
            float g = acc[r] * scale;
            int node = (f < 8) ? sSrc[m] : sTgt[m];
            int c = f & 7;
            atomicAdd(&out[(size_t)(b * NN + node) * CC + c], g);
        }
    }
}

extern "C" void kernel_launch(void* const* d_in, const int* in_sizes, int n_in,
                              void* d_out, int out_size, void* d_ws, size_t ws_size,
                              hipStream_t stream) {
    const float* x  = (const float*)d_in[0];
    const int*   t  = (const int*)d_in[1];
    const int*   xi = (const int*)d_in[2];
    const float* ac = (const float*)d_in[3];
    const float* W1 = (const float*)d_in[4];
    const float* b1 = (const float*)d_in[5];
    const float* W2 = (const float*)d_in[6];
    const float* b2 = (const float*)d_in[7];
    const float* W3 = (const float*)d_in[8];
    float* out = (float*)d_out;

    hipLaunchKernelGGL(zero_kernel, dim3((out_size + 255) / 256), dim3(256), 0, stream,
                       out, out_size);

    dim3 grid(EE / TILE_M, BB);
    hipLaunchKernelGGL(ebm_grad_kernel, grid, dim3(256), 0, stream,
                       x, t, xi, ac, W1, b1, W2, b2, W3, out);
}
